// NNMFBlock2_17978733101074
// MI455X (gfx1250) — matrix operand order repaired and checked by reading, not compiled, not run on hardware
//
#include <hip/hip_runtime.h>
#include <hip/hip_bf16.h>

// ---------------------------------------------------------------------------
// NNMF block for MI455X (gfx1250): bf16 WMMA GEMMs + fused normalizations.
// B=8, C=256, H=W=28, EC=512, 20 iterations.
// ---------------------------------------------------------------------------

typedef __attribute__((ext_vector_type(16))) __bf16 v16bf;
typedef __attribute__((ext_vector_type(8)))  __bf16 v8bf;
typedef __attribute__((ext_vector_type(8)))  float  v8f;

#define B_    8
#define C_    256
#define EC_   512
#define W_    28
#define HW_   784
#define NPIX_ 6272   // B*H*W ; 6272/16 = 392 tiles, 784 % 16 == 0

__device__ __forceinline__ __bf16 f2bf(float f) {
    unsigned u = __float_as_uint(f);
    unsigned r = (u + 0x7FFFu + ((u >> 16) & 1u)) >> 16;   // RNE
    unsigned short s = (unsigned short)r;
    return *(__bf16*)&s;
}

__device__ __forceinline__ v8f wmma_bf16(v16bf a, v16bf b, v8f c) {
    // v_wmma_f32_16x16x32_bf16
    return __builtin_amdgcn_wmma_f32_16x16x32_bf16(
        /*neg_a=*/false, a, /*neg_b=*/false, b,
        /*c_mod=*/(short)0, c, /*reuse_a=*/false, /*reuse_b=*/false);
}

// A-operand load (contiguous k order, matching packed B): lane<16 holds K {0..15},
// lane>=16 holds K {16..31} of row m = lane&15 -> two 16B chunks.
__device__ __forceinline__ v16bf load_A(const __bf16* __restrict__ arow,
                                        int kt, int lane) {
    int c1 = (lane < 16) ? 0 : 16;
    const __bf16* p = arow + kt * 32 + c1;
    v8bf lo = *(const v8bf*)(p);
    v8bf hi = *(const v8bf*)(p + 8);
    return __builtin_shufflevector(lo, hi, 0,1,2,3,4,5,6,7,8,9,10,11,12,13,14,15);
}

// ---------------------------------------------------------------------------
// Weight repack into B-operand lane layout: flat = ((nt*KT+kt)*32+lane)*16+e
// lane: n = nt*16 + (lane&15); K = kt*32 + (lane<16?0:16) + e
// ---------------------------------------------------------------------------
__global__ __launch_bounds__(256) void repack_kernel(
    const float* __restrict__ pw,    // pwconv_w (512,256) row-major
    const float* __restrict__ pw2,   // pw2_w    (256,512) row-major
    __bf16* __restrict__ packR,      // recon B: K=512,N=256
    __bf16* __restrict__ packF,      // fwd   B: K=256,N=512 (= pw^T)
    __bf16* __restrict__ packP)      // pw2   B: K=512,N=256 (= pw2^T)
{
    int idx   = blockIdx.x * 256 + threadIdx.x;     // 0 .. 3*131072-1
    int which = idx >> 17;
    int o     = idx & 131071;
    int e     = o & 15;
    int lane  = (o >> 4) & 31;
    int li    = lane & 15;
    int koff  = (lane < 16) ? 0 : 16;
    if (which == 0) {            // KT=16 (K=512), NT=16 (N=256)
        int kt = (o >> 9) & 15, nt = o >> 13;
        int k = kt * 32 + koff + e, n = nt * 16 + li;
        packR[o] = f2bf(pw[k * 256 + n]);
    } else if (which == 1) {     // KT=8 (K=256), NT=32 (N=512): B[k][n]=pw[n][k]
        int kt = (o >> 9) & 7, nt = o >> 12;
        int k = kt * 32 + koff + e, n = nt * 16 + li;
        packF[o] = f2bf(pw[n * 256 + k]);
    } else {                     // KT=16 (K=512), NT=16 (N=256): B[k][n]=pw2[n][k]
        int kt = (o >> 9) & 15, nt = o >> 13;
        int k = kt * 32 + koff + e, n = nt * 16 + li;
        packP[o] = f2bf(pw2[n * 512 + k]);
    }
}

__global__ void zero_kernel(float* __restrict__ p, int n) {
    int i = blockIdx.x * blockDim.x + threadIdx.x;
    if (i < n) p[i] = 0.f;
}

// per-batch sum of clip(x,1e-6): 196 blocks of 1024 elems per batch image
__global__ __launch_bounds__(256) void sumx_kernel(
    const float* __restrict__ x, float* __restrict__ Sx)
{
    __shared__ float red[8];
    int b = blockIdx.x / 196, blk = blockIdx.x % 196, t = threadIdx.x;
    size_t base = (size_t)b * C_ * HW_ + (size_t)blk * 1024;
    float s = 0.f;
#pragma unroll
    for (int j = 0; j < 4; ++j) s += fmaxf(x[base + t + 256 * j], 1e-6f);
#pragma unroll
    for (int m = 16; m >= 1; m >>= 1) s += __shfl_xor(s, m);
    if ((t & 31) == 0) red[t >> 5] = s;
    __syncthreads();
    if (t == 0) {
        float tot = 0.f;
        for (int j = 0; j < 8; ++j) tot += red[j];
        atomicAdd(&Sx[b], tot);
    }
}

__global__ void xn_kernel(const float* __restrict__ x,
                          const float* __restrict__ Sx,
                          float* __restrict__ xn, int n)
{
    int i = blockIdx.x * 256 + threadIdx.x;
    if (i < n) {
        int b = i / (C_ * HW_);
        xn[i] = fmaxf(x[i], 1e-6f) / fmaxf(Sx[b], 1e-12f);
    }
}

__global__ void hinit_kernel(float* __restrict__ hf, __bf16* __restrict__ hbf, int n) {
    int i = blockIdx.x * 256 + threadIdx.x;
    if (i < n) { hf[i] = 1.0f / 512.0f; hbf[i] = f2bf(1.0f / 512.0f); }
}

// ---------------------------------------------------------------------------
// recon GEMM: z = h(6272x512)bf16 @ packR -> per-pixel L1 norm over C=256
// -> transposed write to z_nchw.  Block = 8 waves, wave handles 2 n-tiles.
// ---------------------------------------------------------------------------
__global__ __launch_bounds__(256) void recon_gemm_kernel(
    const __bf16* __restrict__ hbf, const __bf16* __restrict__ packR,
    float* __restrict__ z_nchw)
{
    __shared__ float zs[16][256];
    __shared__ float psum[16][16];
    __shared__ float rinv[16];
    int tid = threadIdx.x, lane = tid & 31, wv = tid >> 5, tile = blockIdx.x;
    int m = tile * 16 + (lane & 15);
    const __bf16* arow = hbf + (size_t)m * EC_;
    v8f acc0 = {}, acc1 = {};
    int nt0 = wv * 2;
    for (int kt = 0; kt < 16; ++kt) {
        v16bf a  = load_A(arow, kt, lane);
        v16bf b0 = *(const v16bf*)(packR + (size_t)((nt0 * 16 + kt) * 32 + lane) * 16);
        v16bf b1 = *(const v16bf*)(packR + (size_t)(((nt0 + 1) * 16 + kt) * 32 + lane) * 16);
        acc0 = wmma_bf16(a, b0, acc0);
        acc1 = wmma_bf16(a, b1, acc1);
    }
    int row0 = (lane >= 16) ? 8 : 0, col0 = nt0 * 16 + (lane & 15);
#pragma unroll
    for (int r = 0; r < 8; ++r) {
        zs[row0 + r][col0]      = acc0[r];
        zs[row0 + r][col0 + 16] = acc1[r];
    }
    __syncthreads();
    int row = tid >> 4, li = tid & 15;
    float s = 0.f;
#pragma unroll
    for (int j = 0; j < 16; ++j) s += fabsf(zs[row][li + 16 * j]);
    psum[row][li] = s;
    __syncthreads();
    if (tid < 16) {
        float tot = 0.f;
#pragma unroll
        for (int j = 0; j < 16; ++j) tot += psum[tid][j];
        rinv[tid] = 1.0f / fmaxf(tot, 1e-12f);
    }
    __syncthreads();
    // transposed NCHW write: thread = channel c, 16 consecutive pixels
    int c = tid, p0 = tile * 16, b = p0 / HW_, rem = p0 % HW_;
    float* out = z_nchw + ((size_t)(b * C_ + c)) * HW_ + rem;
#pragma unroll
    for (int g = 0; g < 4; ++g) {
        float4 v;
        v.x = zs[4 * g + 0][c] * rinv[4 * g + 0];
        v.y = zs[4 * g + 1][c] * rinv[4 * g + 1];
        v.z = zs[4 * g + 2][c] * rinv[4 * g + 2];
        v.w = zs[4 * g + 3][c] * rinv[4 * g + 3];
        *(float4*)(out + 4 * g) = v;
    }
}

// ---------------------------------------------------------------------------
// depthwise 7x7, pad 3 (one block per (b,c) plane; 34x34 LDS halo tile).
// flip: use reversed kernel (recon path). nhwc_out: write (p,c) layout.
// doclip: clip(.,1e-5). Sacc: per-batch atomic sum of outputs (recon path).
// ---------------------------------------------------------------------------
__global__ __launch_bounds__(256) void dwconv_kernel(
    const float* __restrict__ in_nchw, const float* __restrict__ w49,
    float* __restrict__ out, float* __restrict__ Sacc,
    int flip, int nhwc_out, int doclip)
{
    __shared__ float tile[34 * 34];
    __shared__ float wt[49];
    __shared__ float red[8];
    int t = threadIdx.x, plane = blockIdx.x;
    int b = plane / C_, c = plane % C_;
    const float* ip = in_nchw + (size_t)plane * HW_;
    for (int i = t; i < 34 * 34; i += 256) tile[i] = 0.f;
    __syncthreads();
    for (int i = t; i < HW_; i += 256) {
        int y = i / W_, x = i % W_;
        tile[(y + 3) * 34 + (x + 3)] = ip[i];
    }
    if (t < 49) wt[t] = w49[c * 49 + (flip ? 48 - t : t)];
    __syncthreads();
    float lsum = 0.f;
    for (int i = t; i < HW_; i += 256) {
        int oy = i / W_, ox = i % W_;
        float acc = 0.f;
#pragma unroll
        for (int ky = 0; ky < 7; ++ky)
#pragma unroll
            for (int kx = 0; kx < 7; ++kx)
                acc += tile[(oy + ky) * 34 + ox + kx] * wt[ky * 7 + kx];
        if (doclip) acc = fmaxf(acc, 1e-5f);
        if (nhwc_out) out[((size_t)(b * HW_ + i)) * C_ + c] = acc;
        else          out[(size_t)plane * HW_ + i] = acc;
        lsum += acc;
    }
    if (Sacc) {
#pragma unroll
        for (int m = 16; m >= 1; m >>= 1) lsum += __shfl_xor(lsum, m);
        if ((t & 31) == 0) red[t >> 5] = lsum;
        __syncthreads();
        if (t == 0) {
            float s = 0.f;
            for (int j = 0; j < 8; ++j) s += red[j];
            atomicAdd(&Sacc[b], s);
        }
    }
}

// ratio = xn * Sr[b] / r   (l1norm(r) over (C,H,W) folded in: r/Sr)
__global__ void ratio_kernel(const float* __restrict__ xn,
                             const float* __restrict__ r,
                             const float* __restrict__ Sr,
                             float* __restrict__ ratio, int n)
{
    int i = blockIdx.x * 256 + threadIdx.x;
    if (i < n) {
        int b = i / (C_ * HW_);
        ratio[i] = xn[i] * Sr[b] / r[i];
    }
}

// per-pixel L1 norm over C=256 + bf16 convert; one pixel per wave32
__global__ __launch_bounds__(256) void rownorm_kernel(
    const float* __restrict__ t_nhwc, __bf16* __restrict__ zn)
{
    int lane = threadIdx.x & 31, wv = threadIdx.x >> 5;
    int p = blockIdx.x * 8 + wv;
    const float* row = t_nhwc + (size_t)p * C_;
    float v[8]; float s = 0.f;
#pragma unroll
    for (int j = 0; j < 8; ++j) { v[j] = row[lane + 32 * j]; s += fabsf(v[j]); }
#pragma unroll
    for (int m = 16; m >= 1; m >>= 1) s += __shfl_xor(s, m);
    float inv = 1.0f / fmaxf(s, 1e-12f);
    __bf16* o = zn + (size_t)p * C_;
#pragma unroll
    for (int j = 0; j < 8; ++j) o[lane + 32 * j] = f2bf(v[j] * inv);
}

// ---------------------------------------------------------------------------
// fwd GEMM: u = zn(6272x256) @ packF (N=512) fused with
// h = l1norm(clip(h*u, 1e-5), -1); writes h f32 + bf16.
// Block = 8 waves, wave handles 4 n-tiles (32 acc VGPRs).
// ---------------------------------------------------------------------------
__global__ __launch_bounds__(256) void fwd_gemm_kernel(
    const __bf16* __restrict__ znbf, const __bf16* __restrict__ packF,
    float* __restrict__ hf, __bf16* __restrict__ hbf)
{
    __shared__ float us[16][512];
    __shared__ float psum[16][16];
    __shared__ float rinv[16];
    int tid = threadIdx.x, lane = tid & 31, wv = tid >> 5, tile = blockIdx.x;
    int m = tile * 16 + (lane & 15);
    const __bf16* arow = znbf + (size_t)m * C_;
    v8f acc[4] = {};
    int nt0 = wv * 4;
    for (int kt = 0; kt < 8; ++kt) {
        v16bf a = load_A(arow, kt, lane);
#pragma unroll
        for (int q = 0; q < 4; ++q) {
            v16bf bq = *(const v16bf*)(packF + (size_t)(((nt0 + q) * 8 + kt) * 32 + lane) * 16);
            acc[q] = wmma_bf16(a, bq, acc[q]);
        }
    }
    int row0 = (lane >= 16) ? 8 : 0, col0 = nt0 * 16 + (lane & 15);
#pragma unroll
    for (int q = 0; q < 4; ++q)
#pragma unroll
        for (int r = 0; r < 8; ++r)
            us[row0 + r][col0 + q * 16] = acc[q][r];
    __syncthreads();
    // h*u, clip, partial row sums (thread owns 32 cols of one row)
    int row = tid >> 4, cb = (tid & 15) * 32;
    int p = tile * 16 + row;
    float* hrow = hf + (size_t)p * EC_;
    float s = 0.f;
#pragma unroll
    for (int j = 0; j < 32; ++j) {
        float v = fmaxf(hrow[cb + j] * us[row][cb + j], 1e-5f);
        us[row][cb + j] = v;
        s += v;
    }
    psum[row][tid & 15] = s;
    __syncthreads();
    if (tid < 16) {
        float tot = 0.f;
#pragma unroll
        for (int j = 0; j < 16; ++j) tot += psum[tid][j];
        rinv[tid] = 1.0f / fmaxf(tot, 1e-12f);
    }
    __syncthreads();
    float inv = rinv[row];
    __bf16* hbrow = hbf + (size_t)p * EC_;
#pragma unroll
    for (int j = 0; j < 32; ++j) {
        float v = us[row][cb + j] * inv;
        hrow[cb + j]  = v;
        hbrow[cb + j] = f2bf(v);
    }
}

// ---------------------------------------------------------------------------
// final: y = gamma*(h @ pw2^T + b); out = clip(x,1e-6) + y  (NCHW)
// ---------------------------------------------------------------------------
__global__ __launch_bounds__(256) void final_gemm_kernel(
    const __bf16* __restrict__ hbf, const __bf16* __restrict__ packP,
    const float* __restrict__ x, const float* __restrict__ pw2b,
    const float* __restrict__ gamma, float* __restrict__ out)
{
    __shared__ float zs[16][256];
    int tid = threadIdx.x, lane = tid & 31, wv = tid >> 5, tile = blockIdx.x;
    int m = tile * 16 + (lane & 15);
    const __bf16* arow = hbf + (size_t)m * EC_;
    v8f acc0 = {}, acc1 = {};
    int nt0 = wv * 2;
    for (int kt = 0; kt < 16; ++kt) {
        v16bf a  = load_A(arow, kt, lane);
        v16bf b0 = *(const v16bf*)(packP + (size_t)((nt0 * 16 + kt) * 32 + lane) * 16);
        v16bf b1 = *(const v16bf*)(packP + (size_t)(((nt0 + 1) * 16 + kt) * 32 + lane) * 16);
        acc0 = wmma_bf16(a, b0, acc0);
        acc1 = wmma_bf16(a, b1, acc1);
    }
    int row0 = (lane >= 16) ? 8 : 0, col0 = nt0 * 16 + (lane & 15);
#pragma unroll
    for (int r = 0; r < 8; ++r) {
        zs[row0 + r][col0]      = acc0[r];
        zs[row0 + r][col0 + 16] = acc1[r];
    }
    __syncthreads();
    int c = tid, p0 = tile * 16, b = p0 / HW_, rem = p0 % HW_;
    float g = gamma[c], bias = pw2b[c];
    size_t base = ((size_t)(b * C_ + c)) * HW_ + rem;
#pragma unroll
    for (int q = 0; q < 4; ++q) {
        float4 xv = *(const float4*)(x + base + 4 * q);
        float4 o;
        o.x = fmaxf(xv.x, 1e-6f) + g * (zs[4 * q + 0][c] + bias);
        o.y = fmaxf(xv.y, 1e-6f) + g * (zs[4 * q + 1][c] + bias);
        o.z = fmaxf(xv.z, 1e-6f) + g * (zs[4 * q + 2][c] + bias);
        o.w = fmaxf(xv.w, 1e-6f) + g * (zs[4 * q + 3][c] + bias);
        *(float4*)(out + base + 4 * q) = o;
    }
}

// ---------------------------------------------------------------------------
extern "C" void kernel_launch(void* const* d_in, const int* in_sizes, int n_in,
                              void* d_out, int out_size, void* d_ws, size_t ws_size,
                              hipStream_t stream) {
    (void)in_sizes; (void)n_in; (void)out_size; (void)ws_size;
    const float* x     = (const float*)d_in[0];  // (8,256,28,28)
    const float* dwc_w = (const float*)d_in[1];  // (256,1,7,7)
    const float* pwc_w = (const float*)d_in[2];  // (512,256)
    const float* pw2_w = (const float*)d_in[3];  // (256,512)
    const float* pw2_b = (const float*)d_in[4];  // (256,)
    const float* gamma = (const float*)d_in[5];  // (256,)
    float* out = (float*)d_out;

    char* ws = (char*)d_ws;
    size_t off = 0;
    auto alloc = [&](size_t bytes) -> char* {
        char* p = ws + off; off += (bytes + 255) & ~(size_t)255; return p;
    };
    float*  h_f32  = (float*)  alloc((size_t)NPIX_ * EC_ * 4);  // 12.8 MB
    __bf16* h_bf16 = (__bf16*) alloc((size_t)NPIX_ * EC_ * 2);  //  6.4 MB
    __bf16* zn2_bf = (__bf16*) alloc((size_t)NPIX_ * C_ * 2);   //  3.2 MB
    float*  z_nchw = (float*)  alloc((size_t)B_ * C_ * HW_ * 4);
    float*  r_nchw = (float*)  alloc((size_t)B_ * C_ * HW_ * 4);
    float*  ratio  = (float*)  alloc((size_t)B_ * C_ * HW_ * 4);
    float*  t_nhwc = (float*)  alloc((size_t)NPIX_ * C_ * 4);
    float*  xn     = (float*)  alloc((size_t)B_ * C_ * HW_ * 4);
    __bf16* packR  = (__bf16*) alloc(131072 * 2);
    __bf16* packF  = (__bf16*) alloc(131072 * 2);
    __bf16* packP  = (__bf16*) alloc(131072 * 2);
    float*  Sx     = (float*)  alloc(64);
    float*  Sr     = Sx + 8;

    const int NTOT = B_ * C_ * HW_;

    repack_kernel<<<(3 * 131072) / 256, 256, 0, stream>>>(pwc_w, pw2_w, packR, packF, packP);
    zero_kernel<<<1, 64, 0, stream>>>(Sx, 16);                      // Sx + Sr
    sumx_kernel<<<B_ * 196, 256, 0, stream>>>(x, Sx);
    xn_kernel<<<(NTOT + 255) / 256, 256, 0, stream>>>(x, Sx, xn, NTOT);
    hinit_kernel<<<((NPIX_ * EC_) + 255) / 256, 256, 0, stream>>>(h_f32, h_bf16, NPIX_ * EC_);

    for (int it = 0; it < 20; ++it) {
        recon_gemm_kernel<<<392, 256, 0, stream>>>(h_bf16, packR, z_nchw);
        zero_kernel<<<1, 64, 0, stream>>>(Sr, 8);
        dwconv_kernel<<<B_ * C_, 256, 0, stream>>>(z_nchw, dwc_w, r_nchw, Sr,
                                                   /*flip=*/1, /*nhwc=*/0, /*clip=*/1);
        ratio_kernel<<<(NTOT + 255) / 256, 256, 0, stream>>>(xn, r_nchw, Sr, ratio, NTOT);
        dwconv_kernel<<<B_ * C_, 256, 0, stream>>>(ratio, dwc_w, t_nhwc, nullptr,
                                                   /*flip=*/0, /*nhwc=*/1, /*clip=*/0);
        rownorm_kernel<<<NPIX_ / 8, 256, 0, stream>>>(t_nhwc, zn2_bf);
        fwd_gemm_kernel<<<392, 256, 0, stream>>>(zn2_bf, packF, h_f32, h_bf16);
    }
    final_gemm_kernel<<<392, 256, 0, stream>>>(h_bf16, packP, x, pw2_b, gamma, out);
}
